// Encoder_88802743812519
// MI455X (gfx1250) — matrix operand order repaired and checked by reading, not compiled, not run on hardware
//
#include <hip/hip_runtime.h>

// ---------------------------------------------------------------------------
// Problem constants (B, S, D, H from the reference)
// ---------------------------------------------------------------------------
constexpr int B_ = 2;
constexpr int S_ = 2048;
constexpr int D_ = 256;
constexpr int H_ = 8;
constexpr int HD_ = H_ * D_;   // 2048

typedef __bf16 bf16;
typedef __attribute__((ext_vector_type(16))) __bf16 v16bf;
typedef __attribute__((ext_vector_type(8)))  float  v8f;
typedef __attribute__((ext_vector_type(4)))  unsigned int v4u;
typedef __attribute__((ext_vector_type(8)))  int v8i;
typedef __attribute__((ext_vector_type(4)))  int v4i;

#define WMMA_BF16(a, b, c) \
  __builtin_amdgcn_wmma_f32_16x16x32_bf16(false, (a), false, (b), (short)0, (c), false, false)

// ---------------------------------------------------------------------------
// WMMA fragment loaders (CDNA5 wave32 layouts, CDNA5 ISA §7.12.2)
//
// A (16x32 bf16): lane m = lane&15 holds row m. VGPR0..7 hold K = kb..kb+15
// (2 per VGPR, contiguous), kb = (lane<16) ? 0 : 16.
// -> element i of the v16bf maps to k = kb+i (this file's k order, shared with B).
//
// B (32x16 bf16), loaded from a TRANSPOSED panel Bt[n][k]: lane n = lane&15
// holds column n; K = kb2+0..kb2+15 contiguous, kb2 = (lane<16) ? 0 : 16.
//
// C/D (16x16 f32): VGPR r -> row r (lanes 0-15) / row r+8 (lanes 16-31),
// col = lane&15.
// ---------------------------------------------------------------------------
__device__ __forceinline__ int a_kidx(int i, int lane) {
  int kb = (lane & 16) ? 16 : 0;
  return kb + i;
}

__device__ __forceinline__ v16bf load_a_frag(const bf16* base, int ld) {
  int lane = threadIdx.x & 31;
  const bf16* p = base + (size_t)(lane & 15) * ld;
  v16bf a;
#pragma unroll
  for (int i = 0; i < 16; ++i) a[i] = p[a_kidx(i, lane)];
  return a;
}

// base points at Bt[n0][k0] of a transposed panel with row stride ld (k-major)
__device__ __forceinline__ v16bf load_bt_frag(const bf16* base, int ld) {
  int lane = threadIdx.x & 31;
  int kb = (lane & 16) ? 16 : 0;
  const bf16* p = base + (size_t)(lane & 15) * ld + kb;
  v16bf b;
#pragma unroll
  for (int i = 0; i < 16; ++i) b[i] = p[i];
  return b;
}

// 16-lane-group reductions (masks < 16 keep the two half-waves independent)
__device__ __forceinline__ float redmax16(float v) {
#pragma unroll
  for (int m = 1; m < 16; m <<= 1) v = fmaxf(v, __shfl_xor(v, m, 32));
  return v;
}
__device__ __forceinline__ float redsum16(float v) {
#pragma unroll
  for (int m = 1; m < 16; m <<= 1) v += __shfl_xor(v, m, 32);
  return v;
}

// ---------------------------------------------------------------------------
// Tensor Data Mover: async 2D tile load global -> LDS (CDNA5 ISA-8).
// D# group0: [1:0]=count(1), [63:32]=lds_addr, [120:64]=global_addr,
//            [127:126]=type(2).
// D# group1: [17:16]=data_size(1 -> 2 bytes), [79:48]=tensor_dim0,
//            [111:80]=tensor_dim1, [127:112]=tile_dim0, [143:128]=tile_dim1,
//            [207:160]=tensor_dim0_stride. All dims in data_size units.
// Issue from one wave only (EXEC ignored; TENSORcnt is per-wave), then
// s_wait_tensorcnt 0 + workgroup barrier before consuming the LDS tile.
// This toolchain exposes the 6-arg builtin:
//   (uint32x4 g0, int32x8 g1, int32x4 g2, int32x4 g3, int32x8 pad, i32 cpol)
// ---------------------------------------------------------------------------
__device__ __forceinline__ void tdm_load_2d(const bf16* lds_dst, const bf16* gsrc,
                                            unsigned rows, unsigned row_elems,
                                            unsigned tensor_rows, unsigned row_stride) {
  unsigned long long ga = (unsigned long long)(uintptr_t)(const void*)gsrc;
  unsigned lds_off = (unsigned)(uintptr_t)(const void*)lds_dst;  // low 32 bits = LDS byte offset
  v4u g0;
  g0[0] = 1u;                                                  // count=1, user descriptor
  g0[1] = lds_off;                                             // lds_addr
  g0[2] = (unsigned)ga;                                        // global_addr[31:0]
  g0[3] = (unsigned)((ga >> 32) & 0x01FFFFFFu) | (2u << 30);   // global_addr[56:32] | type=2
  v8i g1;
  g1[0] = (int)(1u << 16);                                     // data_size = 2 bytes
  g1[1] = (int)((row_elems & 0xFFFFu) << 16);                  // tensor_dim0[15:0]
  g1[2] = (int)(((row_elems >> 16) & 0xFFFFu) |
                ((tensor_rows & 0xFFFFu) << 16));              // td0 hi | tensor_dim1 lo
  g1[3] = (int)(((tensor_rows >> 16) & 0xFFFFu) |
                ((row_elems & 0xFFFFu) << 16));                // td1 hi | tile_dim0
  g1[4] = (int)(rows & 0xFFFFu);                               // tile_dim1 (tile_dim2 = 0)
  g1[5] = (int)row_stride;                                     // tensor_dim0_stride[31:0]
  g1[6] = 0;
  g1[7] = 0;
  v4i zz4;
  zz4[0] = 0; zz4[1] = 0; zz4[2] = 0; zz4[3] = 0;
  v8i zz8;
#pragma unroll
  for (int i = 0; i < 8; ++i) zz8[i] = 0;
  __builtin_amdgcn_tensor_load_to_lds(g0, g1, zz4, zz4, zz8, 0);
}

// ---------------------------------------------------------------------------
// Kernel 0: fp32 -> bf16 cast
// ---------------------------------------------------------------------------
__global__ void cast_f32_bf16(const float* __restrict__ src, bf16* __restrict__ dst, int n) {
  int i = blockIdx.x * blockDim.x + threadIdx.x;
  int stride = gridDim.x * blockDim.x;
  for (; i < n; i += stride) dst[i] = (bf16)src[i];
}

// ---------------------------------------------------------------------------
// Kernel 1: Q/K/V projections. grid.z in [0, 3*B*H); 64x64 C tile per block.
// 4 waves x 16 rows. Weight panel transposed into LDS per 64-wide k block.
// B-fragments preloaded in groups of 4 so ds_loads batch ahead of the WMMAs.
// ---------------------------------------------------------------------------
__global__ void qkv_gemm(const bf16* __restrict__ X,
                         const bf16* __restrict__ Wq, const bf16* __restrict__ Wk,
                         const bf16* __restrict__ Wv,
                         bf16* __restrict__ Q, bf16* __restrict__ K, bf16* __restrict__ V) {
  __shared__ bf16 Wt[64 * 64];  // Wt[c][k]

  int z = blockIdx.z;
  int proj = z / (B_ * H_);
  int bh = z % (B_ * H_);
  int b = bh / H_, h = bh % H_;

  const bf16* A = X + (size_t)b * S_ * D_;
  const bf16* W = (proj == 0 ? Wq : (proj == 1 ? Wk : Wv)) + (size_t)h * D_ * D_;
  bf16* C = (proj == 0 ? Q : (proj == 1 ? K : V)) + (size_t)bh * S_ * D_;

  int c0 = blockIdx.x * 64;
  int r0 = blockIdx.y * 64;
  int wave = threadIdx.x >> 5, lane = threadIdx.x & 31;
  int wr0 = r0 + wave * 16;

  v8f acc[4];
#pragma unroll
  for (int t = 0; t < 4; ++t)
#pragma unroll
    for (int r = 0; r < 8; ++r) acc[t][r] = 0.f;

  for (int k0 = 0; k0 < D_; k0 += 64) {
    __syncthreads();
    for (int idx = threadIdx.x; idx < 64 * 64; idx += blockDim.x) {
      int kk = idx >> 6, cc = idx & 63;
      Wt[cc * 64 + kk] = W[(size_t)(k0 + kk) * D_ + c0 + cc];
    }
    __syncthreads();
#pragma unroll
    for (int ks = 0; ks < 64; ks += 32) {
      v16bf a = load_a_frag(A + (size_t)wr0 * D_ + k0 + ks, D_);
      v16bf bfr[4];
#pragma unroll
      for (int t = 0; t < 4; ++t) bfr[t] = load_bt_frag(&Wt[(t * 16) * 64 + ks], 64);
#pragma unroll
      for (int t = 0; t < 4; ++t) acc[t] = WMMA_BF16(a, bfr[t], acc[t]);
    }
  }

  int hi8 = (lane >> 4) * 8, n = lane & 15;
#pragma unroll
  for (int t = 0; t < 4; ++t)
#pragma unroll
    for (int r = 0; r < 8; ++r)
      C[(size_t)(wr0 + r + hi8) * D_ + c0 + t * 16 + n] = (bf16)acc[t][r];
}

// ---------------------------------------------------------------------------
// Kernel 2: fused causal flash attention, per (b,h) head, 64 query rows per
// workgroup (4 waves x 16 rows), 32-wide key blocks.
//   - Q block (64x256) staged once via TDM
//   - K block (32x256) staged per-iteration via TDM
//   - V block transposed manually into LDS (TDM cannot transpose)
// Writes head-concat layout ret[b][s][h*D + d] directly.
// ---------------------------------------------------------------------------
__global__ void flash_attn(const bf16* __restrict__ Qb, const bf16* __restrict__ Kb,
                           const bf16* __restrict__ Vb, bf16* __restrict__ ret) {
  __shared__ bf16 Qlds[64 * D_];      // 32 KB  [q][k]
  __shared__ bf16 Klds[32 * D_];      // 16 KB  [t][k]
  __shared__ bf16 Vt[D_ * 32];        // 16 KB  [d][t]
  __shared__ bf16 Plds[4][16 * 32];   // per-wave P transpose scratch, 4 KB

  int bh = blockIdx.y;
  int b = bh / H_, h = bh % H_;
  int qbase = blockIdx.x * 64;
  int wave = threadIdx.x >> 5, lane = threadIdx.x & 31;
  int qr0 = qbase + wave * 16;
  int hi8 = (lane >> 4) * 8, n = lane & 15;

  const bf16* Qp = Qb + (size_t)bh * S_ * D_;
  const bf16* Kp = Kb + (size_t)bh * S_ * D_;
  const bf16* Vp = Vb + (size_t)bh * S_ * D_;

  // stage the whole Q block once through the Tensor Data Mover
  if (wave == 0) {
    tdm_load_2d(Qlds, Qp + (size_t)qbase * D_, 64, D_, S_, D_);
    __builtin_amdgcn_s_wait_tensorcnt(0);
  }

  v8f o[16];
#pragma unroll
  for (int t = 0; t < 16; ++t)
#pragma unroll
    for (int r = 0; r < 8; ++r) o[t][r] = 0.f;

  float mrow[8], lrow[8];
#pragma unroll
  for (int r = 0; r < 8; ++r) { mrow[r] = -1e30f; lrow[r] = 0.f; }

  const float sc = 0.0625f;  // 1/sqrt(256)
  int nblocks = (qbase >> 5) + 2;  // uniform across the workgroup; tail blocks mask to 0

  for (int blk = 0; blk < nblocks; ++blk) {
    int t0 = blk * 32;

    __syncthreads();  // previous block's LDS reads complete before restaging

    // K block via TDM (async DMA, TENSORcnt), V block transposed by the WG
    if (wave == 0) tdm_load_2d(Klds, Kp + (size_t)t0 * D_, 32, D_, S_, D_);
    for (int idx = threadIdx.x; idx < 32 * D_; idx += blockDim.x) {
      int d = idx & (D_ - 1);
      int t = idx >> 8;
      Vt[d * 32 + t] = Vp[(size_t)(t0 + t) * D_ + d];
    }
    if (wave == 0) __builtin_amdgcn_s_wait_tensorcnt(0);
    __syncthreads();

    // S = Q K^T over full D=256 (8 k-fragments), two 16-col tiles
    v8f s0, s1;
#pragma unroll
    for (int r = 0; r < 8; ++r) { s0[r] = 0.f; s1[r] = 0.f; }
#pragma unroll
    for (int kf = 0; kf < 8; ++kf) {
      v16bf a  = load_a_frag(&Qlds[(wave * 16) * D_ + kf * 32], D_);
      v16bf b0 = load_bt_frag(&Klds[0 * D_ + kf * 32], D_);
      v16bf b1 = load_bt_frag(&Klds[16 * D_ + kf * 32], D_);
      s0 = WMMA_BF16(a, b0, s0);
      s1 = WMMA_BF16(a, b1, s1);
    }

    // online causal softmax (per-row stats live replicated in 16-lane groups)
    float p0[8], p1[8];
#pragma unroll
    for (int r = 0; r < 8; ++r) {
      int qi = qr0 + r + hi8;
      float v0 = s0[r] * sc; if (t0 + n      > qi) v0 = -1e30f;
      float v1 = s1[r] * sc; if (t0 + 16 + n > qi) v1 = -1e30f;
      float mx = redmax16(fmaxf(v0, v1));
      float mnew = fmaxf(mrow[r], mx);
      float rescale = __expf(mrow[r] - mnew);
      float e0 = __expf(v0 - mnew);
      float e1 = __expf(v1 - mnew);
      lrow[r] = lrow[r] * rescale + redsum16(e0 + e1);
      mrow[r] = mnew;
      p0[r] = e0; p1[r] = e1;
#pragma unroll
      for (int t = 0; t < 16; ++t) o[t][r] *= rescale;
    }

    // P: C-layout -> A-layout via per-wave LDS scratch (DS ops in-order per wave)
    bf16* pl = Plds[wave];
#pragma unroll
    for (int r = 0; r < 8; ++r) {
      int row = r + hi8;
      pl[row * 32 + n]      = (bf16)p0[r];
      pl[row * 32 + 16 + n] = (bf16)p1[r];
    }
    v16bf pa = load_a_frag(pl, 32);

    // O += P V  (16 d-tiles, grouped 4 loads + 4 WMMAs for pipelining)
#pragma unroll
    for (int tg = 0; tg < 4; ++tg) {
      v16bf vb[4];
#pragma unroll
      for (int j = 0; j < 4; ++j) vb[j] = load_bt_frag(&Vt[((tg * 4 + j) * 16) * 32], 32);
#pragma unroll
      for (int j = 0; j < 4; ++j) o[tg * 4 + j] = WMMA_BF16(pa, vb[j], o[tg * 4 + j]);
    }

    // prefetch next V key block into cache (K comes via TDM)
    if (blk + 1 < nblocks) {
      __builtin_prefetch(Vp + (size_t)(t0 + 32) * D_, 0, 3);
    }
  }

  // epilogue: divide by row sums, write head-concat ret[b][s][h*D + d]
  bf16* rp = ret + ((size_t)b * S_ + qr0) * HD_ + (size_t)h * D_;
#pragma unroll
  for (int r = 0; r < 8; ++r) {
    float inv = 1.f / lrow[r];
    int row = r + hi8;
#pragma unroll
    for (int t = 0; t < 16; ++t)
      rp[(size_t)row * HD_ + t * 16 + n] = (bf16)(o[t][r] * inv);
  }
}

// ---------------------------------------------------------------------------
// Kernel 3: output projection out[M,N] = ret[M,K] @ wO[K,N], fp32 out.
// M = B*S = 4096, N = 256, K = 2048.
// ---------------------------------------------------------------------------
__global__ void out_gemm(const bf16* __restrict__ A, const bf16* __restrict__ W,
                         float* __restrict__ C, int M, int N, int Kd) {
  __shared__ bf16 Wt[64 * 64];

  int c0 = blockIdx.x * 64;
  int r0 = blockIdx.y * 64;
  int wave = threadIdx.x >> 5, lane = threadIdx.x & 31;
  int wr0 = r0 + wave * 16;

  v8f acc[4];
#pragma unroll
  for (int t = 0; t < 4; ++t)
#pragma unroll
    for (int r = 0; r < 8; ++r) acc[t][r] = 0.f;

  for (int k0 = 0; k0 < Kd; k0 += 64) {
    __syncthreads();
    for (int idx = threadIdx.x; idx < 64 * 64; idx += blockDim.x) {
      int kk = idx >> 6, cc = idx & 63;
      Wt[cc * 64 + kk] = W[(size_t)(k0 + kk) * N + c0 + cc];
    }
    __syncthreads();
#pragma unroll
    for (int ks = 0; ks < 64; ks += 32) {
      v16bf a = load_a_frag(A + (size_t)wr0 * Kd + k0 + ks, Kd);
      v16bf bfr[4];
#pragma unroll
      for (int t = 0; t < 4; ++t) bfr[t] = load_bt_frag(&Wt[(t * 16) * 64 + ks], 64);
#pragma unroll
      for (int t = 0; t < 4; ++t) acc[t] = WMMA_BF16(a, bfr[t], acc[t]);
    }
  }

  int hi8 = (lane >> 4) * 8, n = lane & 15;
#pragma unroll
  for (int t = 0; t < 4; ++t)
#pragma unroll
    for (int r = 0; r < 8; ++r)
      C[(size_t)(wr0 + r + hi8) * N + c0 + t * 16 + n] = acc[t][r];
}

// ---------------------------------------------------------------------------
// Host launcher
// ---------------------------------------------------------------------------
extern "C" void kernel_launch(void* const* d_in, const int* in_sizes, int n_in,
                              void* d_out, int out_size, void* d_ws, size_t ws_size,
                              hipStream_t stream) {
  (void)in_sizes; (void)n_in; (void)out_size; (void)ws_size;

  const float* X  = (const float*)d_in[0];
  // d_in[1] = timestamp, d_in[6] = theta: dead code in the reference output.
  const float* wQ = (const float*)d_in[2];
  const float* wK = (const float*)d_in[3];
  const float* wV = (const float*)d_in[4];
  const float* wO = (const float*)d_in[5];
  float* out = (float*)d_out;

  char* ws = (char*)d_ws;
  size_t off = 0;
  auto carve = [&](size_t nelem) -> bf16* {
    bf16* p = (bf16*)(ws + off);
    off += ((nelem * sizeof(bf16) + 255) / 256) * 256;
    return p;
  };

  bf16* Xb   = carve((size_t)B_ * S_ * D_);
  bf16* wQb  = carve((size_t)H_ * D_ * D_);
  bf16* wKb  = carve((size_t)H_ * D_ * D_);
  bf16* wVb  = carve((size_t)H_ * D_ * D_);
  bf16* wOb  = carve((size_t)HD_ * D_);
  bf16* Qb   = carve((size_t)B_ * H_ * S_ * D_);
  bf16* Kb   = carve((size_t)B_ * H_ * S_ * D_);
  bf16* Vb   = carve((size_t)B_ * H_ * S_ * D_);
  bf16* retb = carve((size_t)B_ * S_ * HD_);

  cast_f32_bf16<<<512, 256, 0, stream>>>(X,  Xb,  B_ * S_ * D_);
  cast_f32_bf16<<<256, 256, 0, stream>>>(wQ, wQb, H_ * D_ * D_);
  cast_f32_bf16<<<256, 256, 0, stream>>>(wK, wKb, H_ * D_ * D_);
  cast_f32_bf16<<<256, 256, 0, stream>>>(wV, wVb, H_ * D_ * D_);
  cast_f32_bf16<<<256, 256, 0, stream>>>(wO, wOb, HD_ * D_);

  // Q/K/V projections: grid (N/64, S/64, 3*B*H)
  qkv_gemm<<<dim3(D_ / 64, S_ / 64, 3 * B_ * H_), 128, 0, stream>>>(
      Xb, wQb, wKb, wVb, Qb, Kb, Vb);

  // fused causal attention: grid (S/64, B*H)
  flash_attn<<<dim3(S_ / 64, B_ * H_), 128, 0, stream>>>(Qb, Kb, Vb, retb);

  // output projection: M = B*S = 4096 -> grid (256/64, 4096/64)
  out_gemm<<<dim3(D_ / 64, (B_ * S_) / 64, 1), 128, 0, stream>>>(
      retb, wOb, out, B_ * S_, D_, HD_);
}
